// GenomicsGraphEncoder_20066087206938
// MI455X (gfx1250) — matrix operand order repaired and checked by reading, not compiled, not run on hardware
//
#include <hip/hip_runtime.h>
#include <hip/hip_bf16.h>

// ---------------- problem constants ----------------
#define Bq      2
#define Nq      2048
#define HIDq    256
#define HEADSq  4
#define Dq      64
#define POOLq   128
#define Mq      (Bq*Nq)          // 4096 rows
#define NWORDS  (Nq/32)          // 64 mask words per row

typedef __attribute__((ext_vector_type(8)))  float    v8f;
typedef __attribute__((ext_vector_type(8)))  __bf16   v8bf;
typedef __attribute__((ext_vector_type(16))) __bf16   v16bf;
typedef __attribute__((ext_vector_type(4)))  unsigned u32x4;
typedef __attribute__((ext_vector_type(8)))  int      i32x8;
typedef __attribute__((ext_vector_type(4)))  int      i32x4;

// TDM availability (device pass only; falls back to ds_store staging otherwise)
#if defined(__HIP_DEVICE_COMPILE__) && defined(__has_builtin)
#if __has_builtin(__builtin_amdgcn_tensor_load_to_lds) && __has_builtin(__builtin_amdgcn_s_wait_tensorcnt)
#define HAVE_TDM 1
#endif
#endif
#ifndef HAVE_TDM
#define HAVE_TDM 0
#endif

__device__ __forceinline__ v16bf cat16(v8bf lo, v8bf hi) {
  return __builtin_shufflevector(lo, hi, 0,1,2,3,4,5,6,7,8,9,10,11,12,13,14,15);
}

// ---------------- adjacency bit-pack: wave32 ballot -> one u32 word ----------------
__global__ void k_pack_adj(const int* __restrict__ adj, unsigned* __restrict__ bits) {
  int id = blockIdx.x * blockDim.x + threadIdx.x;          // B*N*N threads
  int v = adj[id] > 0;
  unsigned long long m = __ballot(v);                      // wave32: low 32 bits
  if ((threadIdx.x & 31) == 0) bits[id >> 5] = (unsigned)m;
}

// ---------------- weight transpose + f32->bf16: out[n][k] = in[k][n] ----------------
__global__ void k_wprep(const float* __restrict__ in, __bf16* __restrict__ out, int K, int Nc) {
  int id = blockIdx.x * blockDim.x + threadIdx.x;
  if (id >= K * Nc) return;
  int k = id / Nc, n = id - k * Nc;
  out[(size_t)n * K + k] = (__bf16)in[id];
}

// ---------------- input projection: x = relu(expr*w + b) ----------------
__global__ void k_input_proj(const float* __restrict__ expr, const float* __restrict__ w,
                             const float* __restrict__ bias, float* __restrict__ xF,
                             __bf16* __restrict__ xbf) {
  int id = blockIdx.x * blockDim.x + threadIdx.x;          // M*HID
  int bn = id >> 8, c = id & 255;
  float v = fmaf(expr[bn], w[c], bias[c]);
  v = v > 0.f ? v : 0.f;
  xF[id] = v;
  xbf[id] = (__bf16)v;
}

// ---------------- generic bf16 WMMA GEMM: C[M,Nc] = A[M,K] * Bt[Nc,K]^T ----------------
__global__ void k_gemm_bf16(const __bf16* __restrict__ A, const __bf16* __restrict__ Bt,
                            float* __restrict__ C, int Mrows, int K, int Nc) {
  int wid  = (blockIdx.x * blockDim.x + threadIdx.x) >> 5;
  int lane = threadIdx.x & 31;
  int tilesN = Nc >> 4;
  int tm = wid / tilesN, tn = wid - tm * tilesN;
  if (tm * 16 >= Mrows) return;
  int arow = tm * 16 + (lane & 15);
  int bcol = tn * 16 + (lane & 15);
  int aoff = (lane < 16) ? 0 : 8;    // 16-bit A layout: V0-3 = K[o..o+7], V4-7 = K[o+16..o+23]
  int boff = (lane < 16) ? 0 : 8;    // 16-bit B layout: V0-3 = K[o..o+7], V4-7 = K[o+16..o+23]
  const __bf16* Ap = A  + (size_t)arow * K;
  const __bf16* Bp = Bt + (size_t)bcol * K;
  v8f acc = {0.f,0.f,0.f,0.f,0.f,0.f,0.f,0.f};
  #pragma unroll 4
  for (int kb = 0; kb < K; kb += 32) {
    v8bf a0 = *(const v8bf*)(Ap + kb + aoff);
    v8bf a1 = *(const v8bf*)(Ap + kb + aoff + 16);
    v8bf b0 = *(const v8bf*)(Bp + kb + boff);
    v8bf b1 = *(const v8bf*)(Bp + kb + boff + 16);
    acc = __builtin_amdgcn_wmma_f32_16x16x32_bf16(false, cat16(a0,a1),
                                                  false, cat16(b0,b1),
                                                  (short)0, acc, false, false);
  }
  int mbase = tm * 16 + ((lane < 16) ? 0 : 8);             // C layout: VGPR r -> row r / r+8
  #pragma unroll
  for (int r = 0; r < 8; ++r)
    C[(size_t)(mbase + r) * Nc + bcol] = acc[r];
}

// ---------------- per-head src/dst projections -> transposed [B,H,N] ----------------
__global__ void k_srcdst(const float* __restrict__ hF, const float* __restrict__ asrc,
                         const float* __restrict__ adst, float* __restrict__ srcT,
                         float* __restrict__ dstT) {
  int id = blockIdx.x * blockDim.x + threadIdx.x;          // B*N*HEADS
  int h = id & 3, n = (id >> 2) & (Nq - 1), b = id >> 13;
  const float* hp = hF + ((size_t)(b * Nq + n)) * HIDq + h * Dq;
  float s = 0.f, d = 0.f;
  #pragma unroll 8
  for (int k = 0; k < Dq; ++k) {
    float v = hp[k];
    s = fmaf(v, asrc[h * Dq + k], s);
    d = fmaf(v, adst[h * Dq + k], d);
  }
  srcT[(b * HEADSq + h) * Nq + n] = s;
  dstT[(b * HEADSq + h) * Nq + n] = d;
}

// ---------------- hF -> bf16 transposed per head: hT[B,H,64,N] ----------------
__global__ void k_htrans(const float* __restrict__ hF, __bf16* __restrict__ hT) {
  int id = blockIdx.x * blockDim.x + threadIdx.x;          // B*N*HID
  int ch = id & 255, n = (id >> 8) & (Nq - 1), b = id >> 19;
  int h = ch >> 6, c = ch & 63;
  hT[(((size_t)(b * HEADSq + h) * Dq + c) << 11) + n] = (__bf16)hF[id];
}

// ---------------- softmax stats over j: row max + sum(exp) ----------------
__global__ void k_stats(const float* __restrict__ srcT, const float* __restrict__ dstT,
                        const unsigned* __restrict__ bits, float* __restrict__ mT,
                        float* __restrict__ lT) {
  int bh = blockIdx.x >> 3, chunk = blockIdx.x & 7;        // grid = B*H*8, block = 256
  int b = bh >> 2;
  __shared__ float dsh[Nq];
  for (int j = threadIdx.x; j < Nq; j += 256) dsh[j] = dstT[bh * Nq + j];
  __syncthreads();
  int i = chunk * 256 + threadIdx.x;
  float si = srcT[bh * Nq + i];
  const unsigned* rb = bits + ((size_t)(b * Nq + i)) * NWORDS;
  float mmax = -3.0e38f;
  for (int w = 0; w < NWORDS; ++w) {
    unsigned word = rb[w];
    #pragma unroll 8
    for (int t = 0; t < 32; ++t) {
      float lg = si + dsh[w * 32 + t];
      lg = lg >= 0.f ? lg : 0.2f * lg;
      lg = ((word >> t) & 1u) ? lg : -1e9f;
      mmax = fmaxf(mmax, lg);
    }
  }
  float sum = 0.f;
  for (int w = 0; w < NWORDS; ++w) {
    unsigned word = rb[w];
    #pragma unroll 8
    for (int t = 0; t < 32; ++t) {
      float lg = si + dsh[w * 32 + t];
      lg = lg >= 0.f ? lg : 0.2f * lg;
      lg = ((word >> t) & 1u) ? lg : -1e9f;
      sum += __expf(lg - mmax);
    }
  }
  mT[bh * Nq + i] = mmax;
  lT[bh * Nq + i] = sum;
}

// ---------------- TDM stage: 64x32 bf16 panel of hT -> LDS ----------------
// D# per CDNA5 ISA ch.8: 2-D tile, data_size=2B, tile 32(x) x 64(y),
// tensor_dim0 = stride = 2048. count=1, type=2 ("image"), no cluster mask.
// Toolchain builtin is the 6-arg form: (g0, g1, g2, g3, g_extra, cpol).
#if HAVE_TDM
__device__ __forceinline__ void tdm_stage(const __bf16* gsrc, unsigned lds_off) {
  unsigned long long ga = (unsigned long long)(size_t)gsrc;
  u32x4 g0;
  g0[0] = 1u;                                  // count=1, is_restore=0, gather=0
  g0[1] = lds_off;                             // lds_addr (bytes)
  g0[2] = (unsigned)ga;                        // global_addr[31:0]
  g0[3] = ((unsigned)(ga >> 32) & 0x01FFFFFFu) | (2u << 30);  // ga[56:32] | type=2
  i32x8 g1;
  g1[0] = 0x00010000;                          // data_size=1 (2 bytes)
  g1[1] = (int)(2048u << 16);                  // tensor_dim0[15:0]=2048 in [31:16]
  g1[2] = (int)(64u << 16);                    // tensor_dim0 hi=0 | tensor_dim1[15:0]=64
  g1[3] = (int)(32u << 16);                    // tensor_dim1 hi=0 | tile_dim0=32
  g1[4] = 64;                                  // tile_dim1=64, tile_dim2=0
  g1[5] = 2048;                                // tensor_dim0_stride[31:0]
  g1[6] = 0;                                   // stride hi | tensor_dim1_stride lo (unused, 2D)
  g1[7] = 0;
  i32x4 z4 = {0, 0, 0, 0};
  i32x8 z8 = {0, 0, 0, 0, 0, 0, 0, 0};
  __builtin_amdgcn_tensor_load_to_lds(g0, g1, z4, z4, z8, 0);
}
#endif

// ---------------- PV: out[i,d] = sum_j softmax(logit)_ij * h[j,d], fused ELU ----------------
// Block = 8 waves sharing one (batch, head); hT j-panel staged into LDS once per block
// per 32-j step (TDM double-buffered when available), WMMA consumes from LDS.
__global__ void k_pv(const float* __restrict__ srcT, const float* __restrict__ dstT,
                     const float* __restrict__ mT, const float* __restrict__ lT,
                     const unsigned* __restrict__ bits, const __bf16* __restrict__ hT,
                     float* __restrict__ outF, __bf16* __restrict__ outbf) {
  __shared__ __bf16 panel[2][Dq * 32];                     // 2 x 4KB double buffer
  int tid  = threadIdx.x;
  int lane = tid & 31;
  int bh = blockIdx.x >> 4;                                // grid = B*H*16 blocks
  int b  = bh >> 2, h = bh & 3;
  int ti = (blockIdx.x & 15) * 8 + (tid >> 5);             // i tile for this wave
  int i  = ti * 16 + (lane & 15);
  float si   = srcT[bh * Nq + i];
  float mi   = mT[bh * Nq + i];
  float invl = 1.0f / lT[bh * Nq + i];
  const unsigned* rb = bits + ((size_t)(b * Nq + i)) * NWORDS;
  const float*  dp = dstT + bh * Nq;
  const __bf16* hb = hT + (size_t)bh * Dq * Nq;
  int jo = (lane < 16) ? 0 : 16;   // A: lane half h holds k = 16h..16h+15 (contiguous order)
  int ko = (lane < 16) ? 0 : 16;   // B: same contiguous k order as A (k = 16h + s)
  int nloc = lane & 15;
  v8f acc0 = {0.f,0.f,0.f,0.f,0.f,0.f,0.f,0.f};
  v8f acc1 = acc0, acc2 = acc0, acc3 = acc0;

  // prologue: stage first panel
#if HAVE_TDM
  if (tid < 32) tdm_stage(hb, (unsigned)(size_t)&panel[0][0]);
#else
  {
    int c = tid >> 2, part = tid & 3;
    *(v8bf*)(&panel[0][0] + tid * 8) = *(const v8bf*)(hb + (size_t)c * Nq + part * 8);
  }
#endif

  for (int jb = 0; jb < Nq; jb += 32) {
    int p = (jb >> 5) & 1;
#if HAVE_TDM
    if (tid < 32) __builtin_amdgcn_s_wait_tensorcnt(0);
#endif
    __syncthreads();                                       // panel[p] ready for all waves
    if (jb + 32 < Nq) {                                    // stage next into other buffer
#if HAVE_TDM
      if (tid < 32) tdm_stage(hb + jb + 32, (unsigned)(size_t)&panel[p ^ 1][0]);
#else
      int c = tid >> 2, part = tid & 3;
      *(v8bf*)(&panel[p ^ 1][0] + tid * 8) =
          *(const v8bf*)(hb + (size_t)c * Nq + jb + 32 + part * 8);
#endif
    }

    unsigned word = rb[jb >> 5];
    v16bf afrag;
    #pragma unroll
    for (int t = 0; t < 8; ++t) {
      int j1 = jo + t, j2 = jo + 8 + t;
      float l1 = si + dp[jb + j1];
      l1 = l1 >= 0.f ? l1 : 0.2f * l1;
      l1 = ((word >> j1) & 1u) ? l1 : -1e9f;
      float l2 = si + dp[jb + j2];
      l2 = l2 >= 0.f ? l2 : 0.2f * l2;
      l2 = ((word >> j2) & 1u) ? l2 : -1e9f;
      afrag[t]     = (__bf16)(__expf(l1 - mi) * invl);
      afrag[8 + t] = (__bf16)(__expf(l2 - mi) * invl);
    }
    const __bf16* pp = &panel[p][0];
    const __bf16* p0 = pp + (0 * 16 + nloc) * 32 + ko;
    const __bf16* p1 = pp + (1 * 16 + nloc) * 32 + ko;
    const __bf16* p2 = pp + (2 * 16 + nloc) * 32 + ko;
    const __bf16* p3 = pp + (3 * 16 + nloc) * 32 + ko;
    acc0 = __builtin_amdgcn_wmma_f32_16x16x32_bf16(false, afrag, false,
             cat16(*(const v8bf*)p0, *(const v8bf*)(p0 + 8)), (short)0, acc0, false, false);
    acc1 = __builtin_amdgcn_wmma_f32_16x16x32_bf16(false, afrag, false,
             cat16(*(const v8bf*)p1, *(const v8bf*)(p1 + 8)), (short)0, acc1, false, false);
    acc2 = __builtin_amdgcn_wmma_f32_16x16x32_bf16(false, afrag, false,
             cat16(*(const v8bf*)p2, *(const v8bf*)(p2 + 8)), (short)0, acc2, false, false);
    acc3 = __builtin_amdgcn_wmma_f32_16x16x32_bf16(false, afrag, false,
             cat16(*(const v8bf*)p3, *(const v8bf*)(p3 + 8)), (short)0, acc3, false, false);
    __syncthreads();                                       // all reads of panel[p] done
  }

  int mbase = ti * 16 + ((lane < 16) ? 0 : 8);
  v8f accs[4] = {acc0, acc1, acc2, acc3};
  #pragma unroll
  for (int s = 0; s < 4; ++s) {
    int ch = h * Dq + s * 16 + nloc;
    #pragma unroll
    for (int r = 0; r < 8; ++r) {
      float v = accs[s][r];
      v = v > 0.f ? v : (__expf(v) - 1.0f);                // ELU fused
      size_t idx = ((size_t)(b * Nq + mbase + r)) * HIDq + ch;
      outF[idx] = v;
      outbf[idx] = (__bf16)v;
    }
  }
}

// ---------------- pool score: s = tanh(t+b1) . w2 + b2 ----------------
__global__ void k_score(const float* __restrict__ t, const float* __restrict__ b1,
                        const float* __restrict__ w2, const float* __restrict__ b2,
                        float* __restrict__ s) {
  int id = blockIdx.x * blockDim.x + threadIdx.x;          // M
  const float* tp = t + (size_t)id * POOLq;
  float acc = b2[0];
  #pragma unroll 4
  for (int p = 0; p < POOLq; ++p) acc = fmaf(tanhf(tp[p] + b1[p]), w2[p], acc);
  s[id] = acc;
}

// ---------------- pooled = softmax(s) . X ----------------
__global__ void k_pool(const float* __restrict__ s, const float* __restrict__ X,
                       float* __restrict__ out) {
  int b = blockIdx.x;                                       // grid = B, block = 256
  __shared__ float aw[Nq];
  __shared__ float red[256];
  int tid = threadIdx.x;
  float mx = -3.0e38f;
  for (int n = tid; n < Nq; n += 256) mx = fmaxf(mx, s[b * Nq + n]);
  red[tid] = mx; __syncthreads();
  for (int st = 128; st > 0; st >>= 1) {
    if (tid < st) red[tid] = fmaxf(red[tid], red[tid + st]);
    __syncthreads();
  }
  float smax = red[0]; __syncthreads();
  float ls = 0.f;
  for (int n = tid; n < Nq; n += 256) {
    float e = __expf(s[b * Nq + n] - smax);
    aw[n] = e; ls += e;
  }
  red[tid] = ls; __syncthreads();
  for (int st = 128; st > 0; st >>= 1) {
    if (tid < st) red[tid] += red[tid + st];
    __syncthreads();
  }
  float inv = 1.0f / red[0];
  float acc = 0.f;
  for (int n = 0; n < Nq; ++n)
    acc = fmaf(aw[n], X[((size_t)(b * Nq + n)) * HIDq + tid], acc);
  out[b * HIDq + tid] = acc * inv;
}

// ---------------- host orchestration ----------------
extern "C" void kernel_launch(void* const* d_in, const int* in_sizes, int n_in,
                              void* d_out, int out_size, void* d_ws, size_t ws_size,
                              hipStream_t stream) {
  const float* gene_expr = (const float*)d_in[0];
  const int*   gene_adj  = (const int*)d_in[1];
  const float* in_w      = (const float*)d_in[2];
  const float* in_b      = (const float*)d_in[3];
  const float* proj_w    = (const float*)d_in[4];
  const float* attn_src  = (const float*)d_in[5];
  const float* attn_dst  = (const float*)d_in[6];
  const float* pool_w1   = (const float*)d_in[7];
  const float* pool_b1   = (const float*)d_in[8];
  const float* pool_w2   = (const float*)d_in[9];
  const float* pool_b2   = (const float*)d_in[10];
  const float* ip_w1     = (const float*)d_in[11];
  const float* ip_b1     = (const float*)d_in[12];
  const float* ip_w2     = (const float*)d_in[13];
  const float* ip_b2     = (const float*)d_in[14];

  // workspace carve-up (256B aligned)
  char* base = (char*)d_ws;
  size_t off = 0;
  auto carve = [&](size_t bytes) { void* p = base + off; off += (bytes + 255) & ~(size_t)255; return p; };
  float*    xF     = (float*)   carve((size_t)Mq * HIDq * 4);
  __bf16*   cur_bf = (__bf16*)  carve((size_t)Mq * HIDq * 2);
  float*    hF     = (float*)   carve((size_t)Mq * HIDq * 4);
  __bf16*   hT     = (__bf16*)  carve((size_t)Bq * HEADSq * Dq * Nq * 2);
  float*    srcT   = (float*)   carve((size_t)Bq * HEADSq * Nq * 4);
  float*    dstT   = (float*)   carve((size_t)Bq * HEADSq * Nq * 4);
  float*    mT     = (float*)   carve((size_t)Bq * HEADSq * Nq * 4);
  float*    lT     = (float*)   carve((size_t)Bq * HEADSq * Nq * 4);
  float*    tbuf   = (float*)   carve((size_t)Mq * POOLq * 4);
  float*    sbuf   = (float*)   carve((size_t)Mq * 4);
  unsigned* bits   = (unsigned*)carve((size_t)Mq * NWORDS * 4);
  __bf16*   wip1t  = (__bf16*)  carve((size_t)POOLq * HIDq * 2);
  __bf16*   wp1t0  = (__bf16*)  carve((size_t)POOLq * HIDq * 2);
  __bf16*   wp1t1  = (__bf16*)  carve((size_t)POOLq * HIDq * 2);
  __bf16*   wpr0t  = (__bf16*)  carve((size_t)HIDq * HIDq * 2);
  __bf16*   wpr1t  = (__bf16*)  carve((size_t)HIDq * HIDq * 2);

  float* out_raw   = (float*)d_out;                        // [B,HID]
  float* out_node0 = out_raw + Bq * HIDq;                  // [B,N,HID]
  float* out_node1 = out_node0 + (size_t)Mq * HIDq;
  float* out_pool0 = out_node1 + (size_t)Mq * HIDq;
  float* out_pool1 = out_pool0 + Bq * HIDq;

  // 1) pack adjacency to bitmask (ballot, wave32)
  k_pack_adj<<<(Bq * Nq * Nq) / 256, 256, 0, stream>>>(gene_adj, bits);

  // 2) weight prep (transpose + bf16)
  k_wprep<<<(HIDq * POOLq + 255) / 256, 256, 0, stream>>>(ip_w1, wip1t, HIDq, POOLq);
  k_wprep<<<(HIDq * POOLq + 255) / 256, 256, 0, stream>>>(pool_w1, wp1t0, HIDq, POOLq);
  k_wprep<<<(HIDq * POOLq + 255) / 256, 256, 0, stream>>>(pool_w1 + HIDq * POOLq, wp1t1, HIDq, POOLq);
  k_wprep<<<(HIDq * HIDq + 255) / 256, 256, 0, stream>>>(proj_w, wpr0t, HIDq, HIDq);
  k_wprep<<<(HIDq * HIDq + 255) / 256, 256, 0, stream>>>(proj_w + HIDq * HIDq, wpr1t, HIDq, HIDq);

  // 3) input projection
  k_input_proj<<<(Mq * HIDq) / 256, 256, 0, stream>>>(gene_expr, in_w, in_b, xF, cur_bf);

  auto run_gemm = [&](const __bf16* A, const __bf16* Bt, float* C, int Nc) {
    int waves = (Mq / 16) * (Nc / 16);
    k_gemm_bf16<<<(waves * 32) / 256, 256, 0, stream>>>(A, Bt, C, Mq, HIDq, Nc);
  };
  auto run_pool = [&](const __bf16* Xbf, const float* Xf, const __bf16* w1t,
                      const float* b1, const float* w2, const float* b2, float* outp) {
    run_gemm(Xbf, w1t, tbuf, POOLq);
    k_score<<<Mq / 256, 256, 0, stream>>>(tbuf, b1, w2, b2, sbuf);
    k_pool<<<Bq, 256, 0, stream>>>(sbuf, Xf, outp);
  };

  // 4) raw pooled on x
  run_pool(cur_bf, xF, wip1t, ip_b1, ip_w2, ip_b2, out_raw);

  // 5) GAT layers
  const __bf16* projT[2] = {wpr0t, wpr1t};
  const __bf16* poolT[2] = {wp1t0, wp1t1};
  float* nodeOut[2] = {out_node0, out_node1};
  float* poolOut[2] = {out_pool0, out_pool1};
  for (int l = 0; l < 2; ++l) {
    run_gemm(cur_bf, projT[l], hF, HIDq);                              // h = cur @ W
    k_srcdst<<<(Bq * Nq * HEADSq) / 256, 256, 0, stream>>>(
        hF, attn_src + l * HEADSq * Dq, attn_dst + l * HEADSq * Dq, srcT, dstT);
    k_htrans<<<(Mq * HIDq) / 256, 256, 0, stream>>>(hF, hT);
    k_stats<<<Bq * HEADSq * 8, 256, 0, stream>>>(srcT, dstT, bits, mT, lT);
    k_pv<<<Bq * HEADSq * 16, 256, 0, stream>>>(
        srcT, dstT, mT, lT, bits, hT, nodeOut[l], cur_bf);             // fused ELU, bf16 copy
    run_pool(cur_bf, nodeOut[l], poolT[l], pool_b1 + l * POOLq,
             pool_w2 + l * POOLq, pool_b2 + l, poolOut[l]);
  }
}
